// WindowAttention_68925635167048
// MI455X (gfx1250) — matrix operand order repaired and checked by reading, not compiled, not run on hardware
//
#include <hip/hip_runtime.h>
#include <hip/hip_bf16.h>

typedef __attribute__((ext_vector_type(16))) _Float16 v16h;
typedef __attribute__((ext_vector_type(8)))  _Float16 v8h;
typedef __attribute__((ext_vector_type(4)))  _Float16 v4h;
typedef __attribute__((ext_vector_type(8)))  float    v8f;

__device__ __forceinline__ v8f wmma_f16(v16h a, v16h b, v8f c) {
  // D = A(16x32 f16) * B(32x16 f16) + C(16x16 f32)
  return __builtin_amdgcn_wmma_f32_16x16x32_f16(false, a, false, b, (short)0, c,
                                                false, false);
}

// A-fragment (16x32 f16) from a row-major f16 row.
// lane L (0..15): row M=L, K = k0..k0+15
// lane L+16     : row M=L, K = k0+16..k0+31
__device__ __forceinline__ v16h loadA_f16(const _Float16* __restrict__ row, int k0, int khi) {
  v8h lo = *(const v8h*)(row + k0 + khi * 16);
  v8h hi = *(const v8h*)(row + k0 + khi * 16 + 8);
  v16h r;
#pragma unroll
  for (int i = 0; i < 8; ++i) { r[i] = lo[i]; r[8 + i] = hi[i]; }
  return r;
}

// B-fragment (32x16 f16): 16 contiguous f16 (weight row = output column).
// lane n: col N=n, K = chunk + (0..15); lane n+16: K += 16 (caller adds khi*16)
__device__ __forceinline__ v16h loadB16_f16(const _Float16* __restrict__ p) {
  v8h lo = *(const v8h*)p;
  v8h hi = *(const v8h*)(p + 8);
  v16h r;
#pragma unroll
  for (int i = 0; i < 8; ++i) { r[i] = lo[i]; r[8 + i] = hi[i]; }
  return r;
}

// ---------------------------------------------------------------------------
// Kernel 0: one-time fp32 -> f16 conversion (x, qkv_w, proj_w).
// ---------------------------------------------------------------------------
__global__ __launch_bounds__(256)
void cvt_f16_kernel(const float* __restrict__ in, _Float16* __restrict__ out, int n4) {
  int i = blockIdx.x * 256 + threadIdx.x;
  if (i < n4) {
    float4 f = ((const float4*)in)[i];
    v4h o;
    o[0] = (_Float16)f.x; o[1] = (_Float16)f.y;
    o[2] = (_Float16)f.z; o[3] = (_Float16)f.w;
    ((v4h*)out)[i] = o;
  }
}

// ---------------------------------------------------------------------------
// Kernel 1: CPB MLP table (225 positions x 16 heads). One block per position.
// ---------------------------------------------------------------------------
__device__ __forceinline__ float relcoord(int a) {
  float t = ((float)(a - 7) / 7.0f) * 8.0f;
  float s = (t > 0.f) ? 1.f : ((t < 0.f) ? -1.f : 0.f);
  return s * log2f(fabsf(t) + 1.0f) * (1.0f / 3.0f);   // / log2(8)
}

__global__ __launch_bounds__(256)
void cpb_table_kernel(const float* __restrict__ w1, const float* __restrict__ b1,
                      const float* __restrict__ w2, float* __restrict__ tbl) {
  __shared__ float acc[16];
  const int p = blockIdx.x;
  const int i = p / 15, j = p % 15;
  const float c0 = relcoord(i), c1 = relcoord(j);
  if (threadIdx.x < 16) acc[threadIdx.x] = 0.f;
  __syncthreads();
  float partial[16];
#pragma unroll
  for (int hh = 0; hh < 16; ++hh) partial[hh] = 0.f;
  for (int u = threadIdx.x; u < 512; u += 256) {
    float hval = fmaxf(w1[u * 2] * c0 + w1[u * 2 + 1] * c1 + b1[u], 0.f);
#pragma unroll
    for (int hh = 0; hh < 16; ++hh) partial[hh] += w2[hh * 512 + u] * hval;
  }
#pragma unroll
  for (int hh = 0; hh < 16; ++hh) atomicAdd(&acc[hh], partial[hh]);
  __syncthreads();
  if (threadIdx.x < 16) tbl[p * 16 + threadIdx.x] = acc[threadIdx.x];
}

// ---------------------------------------------------------------------------
// Kernel 2: expand 16*sigmoid(table) via rel-pos index into rpb (16x64x64).
// ---------------------------------------------------------------------------
__global__ __launch_bounds__(256)
void rpb_expand_kernel(const float* __restrict__ tbl, float* __restrict__ rpb) {
  int g = blockIdx.x * 256 + threadIdx.x;
  int h = g >> 12;
  int rem = g & 4095;
  int n = rem >> 6, m = rem & 63;
  int dh = (n >> 3) - (m >> 3) + 7;
  int dw = (n & 7) - (m & 7) + 7;
  float val = tbl[(dh * 15 + dw) * 16 + h];
  rpb[g] = 16.0f / (1.0f + expf(-val));
}

// ---------------------------------------------------------------------------
// Kernel 3: QKV GEMM (f16 in, f32 acc). C(32768x1536) = Xh * Wh^T + bias,
// scattered into q/k/v (B, nH, N, hd) f16. 128 threads = 4 waves.
// ---------------------------------------------------------------------------
__global__ __launch_bounds__(128)
void qkv_gemm_kernel(const _Float16* __restrict__ xh, const _Float16* __restrict__ wh,
                     const float* __restrict__ q_bias, const float* __restrict__ v_bias,
                     _Float16* __restrict__ q, _Float16* __restrict__ k,
                     _Float16* __restrict__ v) {
  const int lane = threadIdx.x & 31;
  const int wave = threadIdx.x >> 5;
  const int lrow = lane & 15;
  const int khi  = lane >> 4;
  const int row0 = blockIdx.x * 16;
  const int col0 = blockIdx.y * 256 + wave * 64;

  v8f z = {};
  v8f acc[4] = {z, z, z, z};
  const _Float16* xrow = xh + (size_t)(row0 + lrow) * 512;

  for (int k0 = 0; k0 < 512; k0 += 32) {
    __builtin_prefetch(xrow + k0 + 64, 0, 1);
    v16h a = loadA_f16(xrow, k0, khi);
#pragma unroll
    for (int t = 0; t < 4; ++t) {
      int coln = col0 + t * 16 + lrow;
      v16h bf = loadB16_f16(wh + (size_t)coln * 512 + k0 + khi * 16);
      acc[t] = wmma_f16(a, bf, acc[t]);
    }
  }
#pragma unroll
  for (int t = 0; t < 4; ++t) {
#pragma unroll
    for (int r = 0; r < 8; ++r) {
      int M   = row0 + r + khi * 8;
      int col = col0 + t * 16 + lrow;
      int which = col >> 9;        // 0=q, 1=k, 2=v
      int cm = col & 511;
      float bias = (which == 0) ? q_bias[cm] : ((which == 2) ? v_bias[cm] : 0.f);
      float val = acc[t][r] + bias;
      int hh = cm >> 5, d = cm & 31;
      int b = M >> 6, n = M & 63;
      _Float16* dst = (which == 0) ? q : ((which == 1) ? k : v);
      dst[(((size_t)(b * 16 + hh) * 64 + n) * 32) + d] = (_Float16)val;
    }
  }
}

// ---------------------------------------------------------------------------
// Kernel 4: fused per-(b,h) attention. Uses GLOBAL_LOAD_ASYNC_TO_LDS_B128 to
// stage q/k/v tiles (4KB each, f16) into LDS with no VGPR round-trip
// (ASYNCcnt path), then WMMA for qn@kn^T and attn@V. Dynamic LDS only, so
// the shared segment starts at LDS byte offset 0.
// ---------------------------------------------------------------------------
#define ATTN_LDS_BYTES 33536
__global__ __launch_bounds__(256)
void attn_kernel(const _Float16* __restrict__ qf, const _Float16* __restrict__ kf,
                 const _Float16* __restrict__ vf, const float* __restrict__ logit_scale,
                 const float* __restrict__ pos_emb, const float* __restrict__ rpe_comb,
                 const float* __restrict__ rpb, _Float16* __restrict__ attn_x) {
  extern __shared__ char smem[];
  _Float16* qh = (_Float16*)(smem);            // 4096 B  (64x32 f16)
  _Float16* kh = (_Float16*)(smem + 4096);     // 4096 B
  _Float16* vh = (_Float16*)(smem + 8192);     // 4096 B
  float*    at = (float*)(smem + 12288);       // 16384 B (64x64 f32)
  float*    li = (float*)(smem + 28672);       // 4352 B  (64x17 f32)
  float*    rq = (float*)(smem + 33024);       // 256 B
  float*    rk = (float*)(smem + 33280);       // 256 B

  const int h = blockIdx.x;
  const int b = blockIdx.y;
  const int tid  = threadIdx.x;
  const int lane = tid & 31;
  const int wave = tid >> 5;
  const int lrow = lane & 15;
  const int khi  = lane >> 4;

  const size_t base = (size_t)(b * 16 + h) * 2048;
  // Async global->LDS staging of the three 4KB tiles (one B128 per thread each).
  {
    const _Float16* qg = qf + base;
    const _Float16* kg = kf + base;
    const _Float16* vg = vf + base;
    unsigned voff = (unsigned)tid * 16u;        // byte offset within tile
    unsigned dq = voff;                         // LDS byte addresses (segment @0)
    unsigned dk = 4096u + voff;
    unsigned dv = 8192u + voff;
    asm volatile("global_load_async_to_lds_b128 %0, %1, %2"
                 :: "v"(dq), "v"(voff), "s"(qg) : "memory");
    asm volatile("global_load_async_to_lds_b128 %0, %1, %2"
                 :: "v"(dk), "v"(voff), "s"(kg) : "memory");
    asm volatile("global_load_async_to_lds_b128 %0, %1, %2"
                 :: "v"(dv), "v"(voff), "s"(vg) : "memory");
    asm volatile("s_wait_asynccnt 0x0" ::: "memory");
  }
  __syncthreads();

  if (tid < 64) {                               // q/k row reciprocal norms
    float sq = 0.f, sk = 0.f;
#pragma unroll
    for (int d = 0; d < 32; ++d) {
      float a0 = (float)qh[tid * 32 + d]; sq += a0 * a0;
      float c0 = (float)kh[tid * 32 + d]; sk += c0 * c0;
    }
    rq[tid] = rsqrtf(sq);
    rk[tid] = rsqrtf(sk);
  }
  __syncthreads();

  const float scale = expf(fminf(logit_scale[h], 4.605170186f)); // log(1/0.01)

  // attn = (qn @ kn^T) * scale : 4x4 tiles, 2 per wave. Normalization folded
  // into the fragment loads (per-lane scalar multiplies).
#pragma unroll
  for (int tt = 0; tt < 2; ++tt) {
    int t = wave + tt * 8;
    int m0 = (t >> 2) * 16, n0 = (t & 3) * 16;
    float rsc = rq[m0 + lrow];
    float rkc = rk[n0 + lrow];
    v16h a, bf;
#pragma unroll
    for (int hh = 0; hh < 16; ++hh) {
      int g = hh >> 3, jj = hh & 7;
      a[hh] = (_Float16)((float)qh[(m0 + lrow) * 32 + khi * 16 + g * 8 + jj] * rsc);
    }
#pragma unroll
    for (int hh = 0; hh < 16; ++hh)
      bf[hh] = (_Float16)((float)kh[(n0 + lrow) * 32 + khi * 16 + hh] * rkc);
    v8f c = {};
    c = wmma_f16(a, bf, c);
#pragma unroll
    for (int r = 0; r < 8; ++r)
      at[(m0 + r + khi * 8) * 64 + n0 + lrow] = c[r] * scale;
  }
  __syncthreads();

  // logits_int = q @ pos_emb  (64 x 17)
  for (int o = tid; o < 64 * 17; o += 256) {
    int n = o / 17, p = o % 17;
    float s = 0.f;
#pragma unroll
    for (int d = 0; d < 32; ++d) s += (float)qh[n * 32 + d] * pos_emb[d * 17 + p];
    li[o] = s;
  }
  __syncthreads();

  // CoPE (reverse cumsum of sigmoid gates) + RPE combine + softmax, 1 row/thread
  if (tid < 64) {
    const int n = tid;
    const float w0 = rpe_comb[0], w1 = rpe_comb[1];
    const float* rpbrow = rpb + (size_t)(h * 64 + n) * 64;
    float run = 0.f;
    for (int j = 63; j >= 0; --j) {
      float logit = at[n * 64 + j];
      float g = 1.0f / (1.0f + expf(-logit));
      run += g;
      float pos = fminf(run, 16.0f);
      float pf = floorf(pos);
      int pfi = (int)pf;
      int pci = (int)ceilf(pos);
      if (pci > 16) pci = 16;
      float wfrac = pos - pf;
      float ci = li[n * 17 + pci] * wfrac + li[n * 17 + pfi] * (1.0f - wfrac);
      at[n * 64 + j] = logit + w0 * rpbrow[j] + w1 * ci;
    }
    float mx = -3.4e38f;
    for (int j = 0; j < 64; ++j) mx = fmaxf(mx, at[n * 64 + j]);
    float s = 0.f;
    for (int j = 0; j < 64; ++j) {
      float e = expf(at[n * 64 + j] - mx);
      at[n * 64 + j] = e; s += e;
    }
    float inv = 1.0f / s;
    for (int j = 0; j < 64; ++j) at[n * 64 + j] *= inv;
  }
  __syncthreads();

  // out = attn @ v : 4x2 tiles, one per wave; K=64 -> two WMMA steps.
  // B fragments come straight from the f16 LDS tile (no conversion).
  {
    int m0 = (wave >> 1) * 16, n0 = (wave & 1) * 16;
    v8f c = {};
#pragma unroll
    for (int kk = 0; kk < 2; ++kk) {
      int k0 = kk * 32;
      v16h a, bf;
#pragma unroll
      for (int hh = 0; hh < 16; ++hh) {
        int g = hh >> 3, jj = hh & 7;
        a[hh] = (_Float16)at[(m0 + lrow) * 64 + k0 + khi * 16 + g * 8 + jj];
      }
#pragma unroll
      for (int hh = 0; hh < 16; ++hh)
        bf[hh] = vh[(k0 + khi * 16 + hh) * 32 + n0 + lrow];
      c = wmma_f16(a, bf, c);
    }
#pragma unroll
    for (int r = 0; r < 8; ++r) {
      int M = m0 + r + khi * 8;
      int N = n0 + lrow;
      attn_x[((size_t)(b * 64 + M)) * 512 + h * 32 + N] = (_Float16)c[r];
    }
  }
}

// ---------------------------------------------------------------------------
// Kernel 5: output projection. out(32768x512) = attn_x(f16) * proj_wh^T + b.
// ---------------------------------------------------------------------------
__global__ __launch_bounds__(128)
void proj_gemm_kernel(const _Float16* __restrict__ A, const _Float16* __restrict__ Wh,
                      const float* __restrict__ bias, float* __restrict__ out) {
  const int lane = threadIdx.x & 31;
  const int wave = threadIdx.x >> 5;
  const int lrow = lane & 15;
  const int khi  = lane >> 4;
  const int row0 = blockIdx.x * 16;
  const int col0 = blockIdx.y * 256 + wave * 64;

  v8f z = {};
  v8f acc[4] = {z, z, z, z};
  const _Float16* arow = A + (size_t)(row0 + lrow) * 512;

  for (int k0 = 0; k0 < 512; k0 += 32) {
    __builtin_prefetch(arow + k0 + 64, 0, 1);
    v16h a = loadA_f16(arow, k0, khi);
#pragma unroll
    for (int t = 0; t < 4; ++t) {
      int coln = col0 + t * 16 + lrow;
      v16h bf = loadB16_f16(Wh + (size_t)coln * 512 + k0 + khi * 16);
      acc[t] = wmma_f16(a, bf, acc[t]);
    }
  }
#pragma unroll
  for (int t = 0; t < 4; ++t) {
#pragma unroll
    for (int r = 0; r < 8; ++r) {
      int M   = row0 + r + khi * 8;
      int col = col0 + t * 16 + lrow;
      out[(size_t)M * 512 + col] = acc[t][r] + bias[col];
    }
  }
}

// ---------------------------------------------------------------------------
extern "C" void kernel_launch(void* const* d_in, const int* in_sizes, int n_in,
                              void* d_out, int out_size, void* d_ws, size_t ws_size,
                              hipStream_t stream) {
  (void)in_sizes; (void)n_in; (void)out_size; (void)ws_size;
  const float* x           = (const float*)d_in[0];
  const float* qkv_w       = (const float*)d_in[1];
  const float* q_bias      = (const float*)d_in[2];
  const float* v_bias      = (const float*)d_in[3];
  const float* logit_scale = (const float*)d_in[4];
  const float* cpb_w1      = (const float*)d_in[5];
  const float* cpb_b1      = (const float*)d_in[6];
  const float* cpb_w2      = (const float*)d_in[7];
  const float* pos_emb     = (const float*)d_in[8];
  const float* rpe_comb    = (const float*)d_in[9];
  const float* proj_w      = (const float*)d_in[10];
  const float* proj_b      = (const float*)d_in[11];
  float* out = (float*)d_out;

  // Workspace layout (all f16 intermediates; ~170 MB total)
  const size_t QKV_ELEMS = (size_t)512 * 16 * 64 * 32;   // 16,777,216
  const size_t XW_ELEMS  = (size_t)32768 * 512;          // 16,777,216
  const size_t QKVW_ELEMS = (size_t)1536 * 512;          // 786,432
  const size_t PW_ELEMS   = (size_t)512 * 512;           // 262,144
  _Float16* q      = (_Float16*)d_ws;
  _Float16* k      = q + QKV_ELEMS;
  _Float16* v      = k + QKV_ELEMS;
  _Float16* attn_x = v + QKV_ELEMS;
  _Float16* xh     = attn_x + QKV_ELEMS;
  _Float16* wh     = xh + XW_ELEMS;
  _Float16* pwh    = wh + QKVW_ELEMS;
  float*    tbl    = (float*)(pwh + PW_ELEMS);           // 225*16
  float*    rpb    = tbl + 225 * 16;                     // 16*64*64

  // One-time f32 -> f16 conversions (removes all cvt work from GEMM loops).
  cvt_f16_kernel<<<(int)(XW_ELEMS / 4 + 255) / 256, 256, 0, stream>>>(x, xh, (int)(XW_ELEMS / 4));
  cvt_f16_kernel<<<(int)(QKVW_ELEMS / 4 + 255) / 256, 256, 0, stream>>>(qkv_w, wh, (int)(QKVW_ELEMS / 4));
  cvt_f16_kernel<<<(int)(PW_ELEMS / 4 + 255) / 256, 256, 0, stream>>>(proj_w, pwh, (int)(PW_ELEMS / 4));

  cpb_table_kernel<<<225, 256, 0, stream>>>(cpb_w1, cpb_b1, cpb_w2, tbl);
  rpb_expand_kernel<<<256, 256, 0, stream>>>(tbl, rpb);
  qkv_gemm_kernel<<<dim3(2048, 6), 128, 0, stream>>>(xh, wh, q_bias, v_bias, q, k, v);
  attn_kernel<<<dim3(16, 512), 256, ATTN_LDS_BYTES, stream>>>(q, k, v, logit_scale, pos_emb,
                                                              rpe_comb, rpb, attn_x);
  proj_gemm_kernel<<<dim3(2048, 2), 128, 0, stream>>>(attn_x, pwh, proj_b, out);
}
